// ExpKernelMVHawkesProcess_24661702213940
// MI455X (gfx1250) — compile-verified, matrix operand order checked by reading, not run on hardware
//
#include <hip/hip_runtime.h>
#include <hip/hip_bf16.h>
#include <math.h>

typedef __attribute__((ext_vector_type(16))) _Float16 v16h;
typedef __attribute__((ext_vector_type(8)))  float    v8f;

#define BB 8
#define NN 2048
#define DD 10
#define THREADS 256
#define ROWS_PER_BLOCK 128                   // 8 waves * 16 rows
#define BLKS_PER_BATCH (NN / ROWS_PER_BLOCK) // 16

// workspace layout (in floats)
#define WS_MU     0      // 10
#define WS_ALPHA  16     // 100
#define WS_BETA   128    // 100
#define WS_SUMMU  240    // 1
#define WS_PART   256    // BB * BLKS_PER_BATCH = 128

__device__ __forceinline__ float softplus_f(float x) {
    return fmaxf(x, 0.f) + log1pf(__expf(-fabsf(x)));
}

// ---------------------------------------------------------------------------
// Kernel 1: softplus of parameters into workspace + sum(mu)
// ---------------------------------------------------------------------------
__global__ void hawkes_prep(const float* __restrict__ mu_raw,
                            const float* __restrict__ log_alpha,
                            const float* __restrict__ log_beta,
                            float* __restrict__ ws) {
    const int t = threadIdx.x;
    if (t < DD * DD) {
        ws[WS_ALPHA + t] = softplus_f(log_alpha[t]);
        ws[WS_BETA  + t] = softplus_f(log_beta[t]);
    }
    if (t < DD) ws[WS_MU + t] = softplus_f(mu_raw[t]);
    __syncthreads();
    if (t == 0) {
        float s = 0.f;
        for (int d = 0; d < DD; ++d) s += ws[WS_MU + d];
        ws[WS_SUMMU] = s;
    }
}

// ---------------------------------------------------------------------------
// Kernel 2: main O(N^2) triangular pass.
//   grid.x = BB * BLKS_PER_BATCH, block = 256 (8 waves)
//   wave w owns rows i_base..i_base+15; inner loop folds 16x32 pair tiles
//   with v_wmma_f32_16x16x32_f16 against an all-ones B matrix (row-sum).
//   Loop is split: branch-free full tiles (j0+32 <= i_base) + masked tail.
// ---------------------------------------------------------------------------
__global__ __launch_bounds__(THREADS)
void hawkes_main(const float* __restrict__ time_points,
                 const int*   __restrict__ event_types,
                 const float* __restrict__ ws,
                 const int*   __restrict__ Tptr,
                 float*       __restrict__ partials) {
    __shared__ __align__(16) float t_s[NN];
    __shared__ __align__(16) int   e_s[NN];
    __shared__ float2 coef_s[DD * DD];   // (alpha*beta, beta)
    __shared__ float  alpha_s[DD * DD];
    __shared__ float  beta_s[DD * DD];
    __shared__ float  mu_s[DD];
    __shared__ float  rowsum_s[ROWS_PER_BLOCK];
    __shared__ float  red_s[THREADS];

    const int b    = blockIdx.x / BLKS_PER_BATCH;
    const int iblk = blockIdx.x % BLKS_PER_BATCH;
    const int i0   = iblk * ROWS_PER_BLOCK;
    const int tid  = threadIdx.x;

    const float* tp = time_points + (size_t)b * NN;
    const int*   ep = event_types + (size_t)b * NN;
    for (int idx = tid; idx < NN; idx += THREADS) {
        t_s[idx] = tp[idx];
        e_s[idx] = ep[idx];
    }
    if (tid < DD * DD) {
        const float a = ws[WS_ALPHA + tid];
        const float v = ws[WS_BETA + tid];
        alpha_s[tid] = a;
        beta_s[tid]  = v;
        coef_s[tid]  = make_float2(a * v, v);
    }
    if (tid < DD) mu_s[tid] = ws[WS_MU + tid];
    __syncthreads();

    const int wave   = tid >> 5;
    const int lane   = tid & 31;
    const int row    = lane & 15;          // M index in tile
    const int khalf  = (lane >> 4) * 8;    // lanes 16..31 hold K+8
    const int i_base = i0 + wave * 16;
    const int i      = i_base + row;
    const float ti   = t_s[i];
    const int   ei   = e_s[i];
    const float2* __restrict__ crow = &coef_s[ei * DD];

    v8f c = {};
    v16h ones;
#pragma unroll
    for (int q = 0; q < 16; ++q) ones[q] = (_Float16)1.0f;

    // ---- full (unmasked) tiles: every j in [j0, j0+32) is < i for all rows
    const int full_end = i_base & ~31;
    for (int j0 = 0; j0 < full_end; j0 += 32) {
        const int jA = j0 + khalf;        // k = khalf + 0..7
        const int jB = jA + 16;           // k = khalf + 16..23
        // 32B-aligned contiguous runs of 8 -> 4x ds_load_b128 each for t and e
        const float4 tA0 = *(const float4*)&t_s[jA];
        const float4 tA1 = *(const float4*)&t_s[jA + 4];
        const float4 tB0 = *(const float4*)&t_s[jB];
        const float4 tB1 = *(const float4*)&t_s[jB + 4];
        const int4   eA0 = *(const int4*)&e_s[jA];
        const int4   eA1 = *(const int4*)&e_s[jA + 4];
        const int4   eB0 = *(const int4*)&e_s[jB];
        const int4   eB1 = *(const int4*)&e_s[jB + 4];

        const float tv[16] = { tA0.x, tA0.y, tA0.z, tA0.w, tA1.x, tA1.y, tA1.z, tA1.w,
                               tB0.x, tB0.y, tB0.z, tB0.w, tB1.x, tB1.y, tB1.z, tB1.w };
        const int   gv[16] = { eA0.x, eA0.y, eA0.z, eA0.w, eA1.x, eA1.y, eA1.z, eA1.w,
                               eB0.x, eB0.y, eB0.z, eB0.w, eB1.x, eB1.y, eB1.z, eB1.w };
        float2 cf[16];
#pragma unroll
        for (int q = 0; q < 16; ++q) cf[q] = crow[gv[q]];   // one ds_load_b64 each

        v16h a;
#pragma unroll
        for (int q = 0; q < 16; ++q) {
            const float val = cf[q].x * __expf(cf[q].y * (tv[q] - ti));
            a[q] = (_Float16)val;
        }
        c = __builtin_amdgcn_wmma_f32_16x16x32_f16(
                false, a, false, ones, (short)0, c, false, false);
    }

    // ---- masked tail tiles (at most 2): branchless select, loads always in-bounds
    const int jlimit = i_base + 15;       // need j < i; max i in tile is i_base+15
    for (int j0 = full_end; j0 < jlimit; j0 += 32) {
        v16h a;
#pragma unroll
        for (int q = 0; q < 16; ++q) {
            const int k = (q < 8 ? q : q + 8) + khalf;
            const int j = j0 + k;                   // j <= 2047: always in-bounds
            const float  tj = t_s[j];
            const float2 cf = crow[e_s[j]];
            const float  val = cf.x * __expf(cf.y * (tj - ti));
            a[q] = (_Float16)((j < i) ? val : 0.f); // v_cndmask, no branch
        }
        c = __builtin_amdgcn_wmma_f32_16x16x32_f16(
                false, a, false, ones, (short)0, c, false, false);
    }

    // C/D layout: VGPR r -> (lanes 0..15: M=r, N=lane), (lanes 16..31: M=8+r)
    if (lane == 0) {
#pragma unroll
        for (int r = 0; r < 8; ++r) rowsum_s[wave * 16 + r] = c[r];
    } else if (lane == 16) {
#pragma unroll
        for (int r = 0; r < 8; ++r) rowsum_s[wave * 16 + 8 + r] = c[r];
    }
    __syncthreads();

    const float Tf = (float)(*Tptr);
    float partial = 0.f;
    if (tid < ROWS_PER_BLOCK) {
        const int   gi = i0 + tid;
        const int   eg = e_s[gi];
        const float inten = mu_s[eg] + rowsum_s[tid];
        partial = logf(inten);                 // + log-intensity
        const float delta = Tf - t_s[gi];      // compensator contribution of event gi
        float s = 0.f;
#pragma unroll
        for (int d = 0; d < DD; ++d) {
            const float bv = beta_s[d * DD + eg];
            s += alpha_s[d * DD + eg] * (1.f - __expf(-bv * delta));
        }
        partial -= s;
    }
    red_s[tid] = partial;
    __syncthreads();
    for (int off = THREADS / 2; off > 0; off >>= 1) {
        if (tid < off) red_s[tid] += red_s[tid + off];
        __syncthreads();
    }
    if (tid == 0) partials[blockIdx.x] = red_s[0];
}

// ---------------------------------------------------------------------------
// Kernel 3: deterministic fixed-order combine of per-block partials
// ---------------------------------------------------------------------------
__global__ void hawkes_final(const float* __restrict__ ws,
                             const int*   __restrict__ Tptr,
                             float*       __restrict__ out) {
    const int b = threadIdx.x;
    if (b < BB) {
        const float Tf = (float)(*Tptr);
        float s = -Tf * ws[WS_SUMMU];          // -sum_d mu_d * T
        for (int k = 0; k < BLKS_PER_BATCH; ++k)
            s += ws[WS_PART + b * BLKS_PER_BATCH + k];
        out[b] = s;
    }
}

extern "C" void kernel_launch(void* const* d_in, const int* in_sizes, int n_in,
                              void* d_out, int out_size, void* d_ws, size_t ws_size,
                              hipStream_t stream) {
    (void)in_sizes; (void)n_in; (void)out_size; (void)ws_size;
    const float* time_points = (const float*)d_in[0];
    const int*   event_types = (const int*)d_in[1];
    const float* mu_raw      = (const float*)d_in[2];
    const float* log_alpha   = (const float*)d_in[3];
    const float* log_beta    = (const float*)d_in[4];
    const int*   Tptr        = (const int*)d_in[5];

    float* ws  = (float*)d_ws;
    float* out = (float*)d_out;
    float* partials = ws + WS_PART;

    hawkes_prep<<<1, 128, 0, stream>>>(mu_raw, log_alpha, log_beta, ws);
    hawkes_main<<<BB * BLKS_PER_BATCH, THREADS, 0, stream>>>(
        time_points, event_types, ws, Tptr, partials);
    hawkes_final<<<1, 32, 0, stream>>>(ws, Tptr, out);
}
